// SwinTransformerBlock_42511586296000
// MI455X (gfx1250) — matrix operand order repaired and checked by reading, not compiled, not run on hardware
//
#include <hip/hip_runtime.h>

typedef __attribute__((ext_vector_type(16))) __bf16 v16bf;
typedef __attribute__((ext_vector_type(8)))  __bf16 v8bf;
typedef __attribute__((ext_vector_type(8)))  float  v8f;

#define NWTOT 2048   // 32 images * 64 windows
#define NP    64     // padded tokens per window (49 -> 64)
#define CDIM  192
#define NHEAD 6
#define HD    32
#define NTOK  49
#define TOKS  100352 // 32*56*56
#define C3    576
#define CF    768

// ---------------------------------------------------------------------------
// WMMA fragment loaders (contiguous-K order used by A and B here, not AMD's table order; cf. CDNA5 ISA §7.12.2)
// A 16x32 bf16: lane L holds row M=L%16; one contiguous 16-elem K run:
//   lanes 0-15:  K = 0..15 ; lanes 16-31: K = 16..31
// ---------------------------------------------------------------------------
__device__ inline v16bf load_a_frag(const __bf16* Arow, int lda) {
    int lane = threadIdx.x & 31;
    int m    = lane & 15;
    int kb   = (lane >> 4) << 4;              // 0 or 16
    const __bf16* p = Arow + (size_t)m * lda + kb;
    v8bf lo = *(const v8bf*)(p);
    v8bf hi = *(const v8bf*)(p + 8);
    return __builtin_shufflevector(lo, hi, 0,1,2,3,4,5,6,7,8,9,10,11,12,13,14,15);
}

// B 32x16 bf16 where column n is CONTIGUOUS in memory (i.e. source is W^T
// [N][K] row-major, or K^T from row-major K). colbase points at column 0 of
// the 16-wide tile; ldc = stride between columns.
// lanes 0-15: n=L, K=0..15 ; lanes 16-31: n=L-16, K=16..31
__device__ inline v16bf load_b_frag_ct(const __bf16* colbase, int ldc) {
    int lane = threadIdx.x & 31;
    int n    = lane & 15;
    int kh   = (lane >> 4) << 4;              // 0 or 16
    const __bf16* p = colbase + (size_t)n * ldc + kh;
    v8bf lo = *(const v8bf*)(p);
    v8bf hi = *(const v8bf*)(p + 8);
    return __builtin_shufflevector(lo, hi, 0,1,2,3,4,5,6,7,8,9,10,11,12,13,14,15);
}

// B 32x16 bf16 from row-major K x N source (stride ldb between K rows).
__device__ inline v16bf load_b_frag_rm(const __bf16* Bm, int ldb) {
    int lane = threadIdx.x & 31;
    int n    = lane & 15;
    int kh   = (lane >> 4) << 4;
    v16bf b;
#pragma unroll
    for (int v = 0; v < 8; ++v) {
        int k = kh + 2 * v;
        b[2*v]   = Bm[(size_t)k       * ldb + n];
        b[2*v+1] = Bm[(size_t)(k + 1) * ldb + n];
    }
    return b;
}

// ---------------------------------------------------------------------------
// Weight convert + transpose: f32 [K][N] -> bf16 [N][K]
// ---------------------------------------------------------------------------
__global__ void cvtT_kernel(const float* __restrict__ s, __bf16* __restrict__ d,
                            int K, int N) {
    int i = blockIdx.x * 256 + threadIdx.x;
    if (i < K * N) {
        int k = i / N, n = i % N;
        d[(size_t)n * K + k] = (__bf16)s[i];
    }
}

// ---------------------------------------------------------------------------
// Kernel 1: LayerNorm1 + cyclic shift (-3,-3) + window partition -> bf16
// one block per padded window token; rows 49..63 zero-filled
// ---------------------------------------------------------------------------
__global__ __launch_bounds__(192) void ln1_shift_part_kernel(
    const float* __restrict__ x, const float* __restrict__ w,
    const float* __restrict__ bparm, __bf16* __restrict__ xw) {
    __shared__ float s1[256], s2[256];
    int blk = blockIdx.x;               // win*64 + p
    int c   = threadIdx.x;              // 0..191
    int win = blk >> 6, p = blk & 63;
    if (p >= NTOK) { xw[(size_t)blk * CDIM + c] = (__bf16)0.f; return; }
    int b_ = win >> 6, wi = win & 63, wh = wi >> 3, ww = wi & 7;
    int i = p / 7, j = p % 7;
    int hh = (wh * 7 + i + 3) % 56;     // roll by -SHIFT: src = (dst+3)%56
    int wc = (ww * 7 + j + 3) % 56;
    const float* row = x + (((size_t)b_ * 56 + hh) * 56 + wc) * CDIM;
    float v = row[c];
    s1[c] = v; s2[c] = v * v;
    if (c < 64) { s1[192 + c] = 0.f; s2[192 + c] = 0.f; }
    __syncthreads();
    for (int s = 128; s > 0; s >>= 1) {
        if (c < s) { s1[c] += s1[c + s]; s2[c] += s2[c + s]; }
        __syncthreads();
    }
    float mean = s1[0] * (1.f / 192.f);
    float var  = s2[0] * (1.f / 192.f) - mean * mean;
    float y = (v - mean) * rsqrtf(var + 1e-5f) * w[c] + bparm[c];
    xw[(size_t)blk * CDIM + c] = (__bf16)y;
}

// ---------------------------------------------------------------------------
// Kernel 5: plain LayerNorm2 (x1 f32 -> xn2 bf16)
// ---------------------------------------------------------------------------
__global__ __launch_bounds__(192) void ln2_kernel(
    const float* __restrict__ x1, const float* __restrict__ w,
    const float* __restrict__ bparm, __bf16* __restrict__ xn2) {
    __shared__ float s1[256], s2[256];
    int c = threadIdx.x;
    const float* row = x1 + (size_t)blockIdx.x * CDIM;
    float v = row[c];
    s1[c] = v; s2[c] = v * v;
    if (c < 64) { s1[192 + c] = 0.f; s2[192 + c] = 0.f; }
    __syncthreads();
    for (int s = 128; s > 0; s >>= 1) {
        if (c < s) { s1[c] += s1[c + s]; s2[c] += s2[c + s]; }
        __syncthreads();
    }
    float mean = s1[0] * (1.f / 192.f);
    float var  = s2[0] * (1.f / 192.f) - mean * mean;
    float y = (v - mean) * rsqrtf(var + 1e-5f) * w[c] + bparm[c];
    xn2[(size_t)blockIdx.x * CDIM + c] = (__bf16)y;
}

// ---------------------------------------------------------------------------
// Generic WMMA GEMM: C[M,N] = A[M,K](bf16) * W^T[N,K](bf16) + bias, epilogues:
//  EPI 0: store bf16 (QKV)
//  EPI 1: window-reverse + roll + residual scatter -> x1 f32 (proj)
//  EPI 2: exact GELU, store bf16 (fc1)
//  EPI 3: + residual(aux) -> f32 out (fc2)
// block = 256 thr = 8 waves; wave tile 16x64; block tile 128x64
// ---------------------------------------------------------------------------
template <int EPI>
__global__ __launch_bounds__(256) void gemm_wmma(
    const __bf16* __restrict__ A, int lda,
    const __bf16* __restrict__ BT, int ldb,     // [N][K], ldb = K
    const float* __restrict__ bias,
    void* __restrict__ outp, const float* __restrict__ aux,
    int M, int N, int K) {
    int wave = threadIdx.x >> 5;
    int lane = threadIdx.x & 31;
    int row0 = blockIdx.x * 128 + wave * 16;
    int col0 = blockIdx.y * 64;
    v8f acc[4] = {};
    for (int kk = 0; kk < K; kk += 32) {
        v16bf a = load_a_frag(A + (size_t)row0 * lda + kk, lda);
#pragma unroll
        for (int t = 0; t < 4; ++t) {
            v16bf b = load_b_frag_ct(BT + (size_t)(col0 + t * 16) * ldb + kk, ldb);
            acc[t] = __builtin_amdgcn_wmma_f32_16x16x32_bf16(
                false, a, false, b, (short)0, acc[t], false, false);
        }
    }
    int mlo = (lane >> 4) << 3;
    int n   = lane & 15;
#pragma unroll
    for (int t = 0; t < 4; ++t) {
        int col = col0 + t * 16 + n;
        float bv = bias[col];
#pragma unroll
        for (int r = 0; r < 8; ++r) {
            int row = row0 + mlo + r;
            float val = acc[t][r] + bv;
            if (EPI == 0) {
                ((__bf16*)outp)[(size_t)row * N + col] = (__bf16)val;
            } else if (EPI == 1) {
                int win = row >> 6, p = row & 63;
                if (p < NTOK) {
                    int b_ = win >> 6, wi = win & 63, wh = wi >> 3, ww = wi & 7;
                    int i = p / 7, j = p % 7;
                    int hh = (wh * 7 + i + 3) % 56;   // reverse + roll(+3)
                    int wc = (ww * 7 + j + 3) % 56;
                    size_t tok = ((size_t)b_ * 56 + hh) * 56 + wc;
                    ((float*)outp)[tok * CDIM + col] = aux[tok * CDIM + col] + val;
                }
            } else if (EPI == 2) {
                float g = 0.5f * val * (1.0f + erff(val * 0.70710678118f));
                ((__bf16*)outp)[(size_t)row * N + col] = (__bf16)g;
            } else {
                ((float*)outp)[(size_t)row * N + col] =
                    val + aux[(size_t)row * N + col];
            }
        }
    }
}

// ---------------------------------------------------------------------------
// Kernel 3: fused window attention, one block (128 thr = 4 waves) per
// (window, head). S = scale*(q k^T) + rel-pos-bias + shift-mask; softmax over
// 49 valid keys; O = P v. Bias/mask computed analytically.
// ---------------------------------------------------------------------------
__global__ __launch_bounds__(128) void attn_kernel(
    const __bf16* __restrict__ qkv,   // [NWTOT*NP][576]
    const float* __restrict__ rpb,    // [169][6]
    __bf16* __restrict__ outO) {      // [NWTOT*NP][192]
    __shared__ __align__(16) float  sS[64 * 66];
    __shared__ __align__(16) __bf16 sP[64 * 72];
    int win = blockIdx.x, head = blockIdx.y;
    int wave = threadIdx.x >> 5, lane = threadIdx.x & 31;
    const __bf16* qp = qkv + (size_t)win * NP * C3 + head * HD;
    const __bf16* kp = qp + CDIM;
    const __bf16* vp = qp + 2 * CDIM;
    int row0 = wave * 16;
    int mlo  = (lane >> 4) << 3;
    int ncol = lane & 15;

    // S = q k^T  (K = HD = 32 -> one WMMA per 16x16 tile)
    v16bf a = load_a_frag(qp + (size_t)row0 * C3, C3);
#pragma unroll
    for (int t = 0; t < 4; ++t) {
        v16bf b = load_b_frag_ct(kp + (size_t)(t * 16) * C3, C3);
        v8f acc = {};
        acc = __builtin_amdgcn_wmma_f32_16x16x32_bf16(
            false, a, false, b, (short)0, acc, false, false);
#pragma unroll
        for (int r = 0; r < 8; ++r)
            sS[(row0 + mlo + r) * 66 + t * 16 + ncol] = acc[r] * 0.17677669529f;
    }
    __syncthreads();

    // softmax rows (64 row-workers)
    int wi = win & 63, wh = wi >> 3, ww = wi & 7;
    if (threadIdx.x < 64) {
        int nr = threadIdx.x;
        if (nr < NTOK) {
            int i1 = nr / 7, j1 = nr % 7;
            int h1 = wh * 7 + i1, w1 = ww * 7 + j1;
            int r1 = (h1 < 49 ? 0 : (h1 < 53 ? 1 : 2)) * 3 +
                     (w1 < 49 ? 0 : (w1 < 53 ? 1 : 2));
            float mx = -1e30f;
            for (int m = 0; m < NTOK; ++m) {
                int i2 = m / 7, j2 = m % 7;
                float bi = rpb[((i1 - i2 + 6) * 13 + (j1 - j2 + 6)) * NHEAD + head];
                int h2 = wh * 7 + i2, w2 = ww * 7 + j2;
                int r2 = (h2 < 49 ? 0 : (h2 < 53 ? 1 : 2)) * 3 +
                         (w2 < 49 ? 0 : (w2 < 53 ? 1 : 2));
                float v = sS[nr * 66 + m] + bi + (r1 == r2 ? 0.f : -100.f);
                sS[nr * 66 + m] = v;
                mx = fmaxf(mx, v);
            }
            float sum = 0.f;
            for (int m = 0; m < NTOK; ++m) {
                float e = __expf(sS[nr * 66 + m] - mx);
                sS[nr * 66 + m] = e;
                sum += e;
            }
            float rs = 1.f / sum;
            for (int m = 0; m < 64; ++m)
                sP[nr * 72 + m] = (__bf16)(m < NTOK ? sS[nr * 66 + m] * rs : 0.f);
        } else {
            for (int m = 0; m < 64; ++m) sP[nr * 72 + m] = (__bf16)0.f;
        }
    }
    __syncthreads();

    // O = P v : wave -> 16 rows x 32 cols (2 tiles), K = 64 in two steps
    v8f acco[2] = {};
#pragma unroll
    for (int kk = 0; kk < 64; kk += 32) {
        v16bf pa = load_a_frag(sP + (size_t)row0 * 72 + kk, 72);
#pragma unroll
        for (int t = 0; t < 2; ++t) {
            v16bf vb = load_b_frag_rm(vp + (size_t)kk * C3 + t * 16, C3);
            acco[t] = __builtin_amdgcn_wmma_f32_16x16x32_bf16(
                false, pa, false, vb, (short)0, acco[t], false, false);
        }
    }
#pragma unroll
    for (int t = 0; t < 2; ++t)
#pragma unroll
        for (int r = 0; r < 8; ++r) {
            int row = row0 + mlo + r;
            outO[((size_t)win * NP + row) * CDIM + head * HD + t * 16 + ncol] =
                (__bf16)acco[t][r];
        }
}

// ---------------------------------------------------------------------------
extern "C" void kernel_launch(void* const* d_in, const int* in_sizes, int n_in,
                              void* d_out, int out_size, void* d_ws,
                              size_t ws_size, hipStream_t stream) {
    const float* x      = (const float*)d_in[0];
    const float* n1w    = (const float*)d_in[1];
    const float* n1b    = (const float*)d_in[2];
    const float* qkv_w  = (const float*)d_in[3];
    const float* qkv_b  = (const float*)d_in[4];
    const float* proj_w = (const float*)d_in[5];
    const float* proj_b = (const float*)d_in[6];
    const float* rpb    = (const float*)d_in[7];
    const float* n2w    = (const float*)d_in[8];
    const float* n2b    = (const float*)d_in[9];
    const float* fc1_w  = (const float*)d_in[10];
    const float* fc1_b  = (const float*)d_in[11];
    const float* fc2_w  = (const float*)d_in[12];
    const float* fc2_b  = (const float*)d_in[13];
    float* out = (float*)d_out;
    char*  ws  = (char*)d_ws;

    // workspace layout (bytes)
    __bf16* wq  = (__bf16*)ws;                 // 192*576 transposed bf16
    __bf16* wp  = wq + 192 * 576;
    __bf16* wf1 = wp + 192 * 192;
    __bf16* wf2 = wf1 + 192 * 768;
    constexpr size_t OFF_XW  = (size_t)1 << 20;
    constexpr size_t OFF_QKV = OFF_XW + (size_t)NWTOT * NP * CDIM * 2;
    constexpr size_t OFF_X1  = OFF_QKV + (size_t)NWTOT * NP * C3 * 2;
    constexpr size_t OFF_XN2 = OFF_X1 + (size_t)TOKS * CDIM * 4;
    __bf16* xw      = (__bf16*)(ws + OFF_XW);
    __bf16* qkvb    = (__bf16*)(ws + OFF_QKV);
    float*  x1      = (float*)(ws + OFF_X1);
    __bf16* xn2     = (__bf16*)(ws + OFF_XN2);
    __bf16* attnout = xw;   // xw dead after QKV GEMM
    __bf16* hbuf    = xw;   // xw+qkv region dead after proj GEMM (154 MB fits)

    // 0) weights -> transposed bf16
    cvtT_kernel<<<(192 * 576 + 255) / 256, 256, 0, stream>>>(qkv_w, wq, 192, 576);
    cvtT_kernel<<<(192 * 192 + 255) / 256, 256, 0, stream>>>(proj_w, wp, 192, 192);
    cvtT_kernel<<<(192 * 768 + 255) / 256, 256, 0, stream>>>(fc1_w, wf1, 192, 768);
    cvtT_kernel<<<(768 * 192 + 255) / 256, 256, 0, stream>>>(fc2_w, wf2, 768, 192);

    // 1) LN1 + shift + partition
    ln1_shift_part_kernel<<<NWTOT * NP, 192, 0, stream>>>(x, n1w, n1b, xw);

    // 2) QKV GEMM: M=131072, N=576, K=192
    gemm_wmma<0><<<dim3(NWTOT * NP / 128, C3 / 64), 256, 0, stream>>>(
        xw, CDIM, wq, CDIM, qkv_b, (void*)qkvb, nullptr, NWTOT * NP, C3, CDIM);

    // 3) fused window attention
    attn_kernel<<<dim3(NWTOT, NHEAD), 128, 0, stream>>>(qkvb, rpb, attnout);

    // 4) proj GEMM + window-reverse + roll + residual -> x1
    gemm_wmma<1><<<dim3(NWTOT * NP / 128, CDIM / 64), 256, 0, stream>>>(
        attnout, CDIM, wp, CDIM, proj_b, (void*)x1, x, NWTOT * NP, CDIM, CDIM);

    // 5) LN2
    ln2_kernel<<<TOKS, 192, 0, stream>>>(x1, n2w, n2b, xn2);

    // 6) FC1 + GELU: M=100352, N=768, K=192
    gemm_wmma<2><<<dim3(TOKS / 128, CF / 64), 256, 0, stream>>>(
        xn2, CDIM, wf1, CDIM, fc1_b, (void*)hbuf, nullptr, TOKS, CF, CDIM);

    // 7) FC2 + residual -> out: M=100352, N=192, K=768
    gemm_wmma<3><<<dim3(TOKS / 128, CDIM / 64), 256, 0, stream>>>(
        hbuf, CF, wf2, CF, fc2_b, (void*)out, x1, TOKS, CDIM, CF);
}
